// WiditBlock_54382875902497
// MI455X (gfx1250) — matrix operand order repaired and checked by reading, not compiled, not run on hardware
//
#include <hip/hip_runtime.h>
#include <hip/hip_bf16.h>

// ---------------- problem constants ----------------
#define NIMG   8
#define TSEQ   4096
#define CDIM   768
#define WSZ    8
#define SHIFTC 4
#define NHEAD  12
#define HD     64
#define HIDD   3072
#define NWIN   64                 // windows per image (8x8)
#define BWIN   (NIMG * NWIN)      // 512
#define TW     64                 // tokens per window
#define MROWS  (BWIN * TW)        // 32768

typedef __attribute__((ext_vector_type(16))) _Float16 v16h;
typedef __attribute__((ext_vector_type(8)))  float    v8f;
typedef __attribute__((ext_vector_type(4)))  int      i4vec;

// ---------------- CDNA5 async global->LDS copy (ASYNCcnt path) ----------------
#if defined(__HIP_DEVICE_COMPILE__) && \
    __has_builtin(__builtin_amdgcn_global_load_async_to_lds_b128) && \
    __has_builtin(__builtin_amdgcn_s_wait_asynccnt)
#define HAVE_ASYNC_LDS 1
#else
#define HAVE_ASYNC_LDS 0
#endif

__device__ __forceinline__ void cp16_async(const _Float16* g, _Float16* l) {
#if HAVE_ASYNC_LDS
    i4vec* gg = (i4vec*)g;   // drop const, reinterpret as 16B vector (generic AS)
    i4vec* ll = (i4vec*)l;
    __builtin_amdgcn_global_load_async_to_lds_b128(
        (__attribute__((address_space(1))) i4vec*)gg,
        (__attribute__((address_space(3))) i4vec*)ll, 0, 0);
#else
    *reinterpret_cast<uint4*>(l) = *reinterpret_cast<const uint4*>(g);
#endif
}

__device__ __forceinline__ void cp_wait_all() {
#if HAVE_ASYNC_LDS
    __builtin_amdgcn_s_wait_asynccnt(0);
#endif
}

// ---------------- WMMA fragment helpers (CDNA5 16x16x32 f16 layout) ----------------
// A (16x32, MxK): lane L -> m = L%16, group g = L/16.
//   a[i]   = A[m][kbase + g*8 + i]        i = 0..7
//   a[8+i] = A[m][kbase + 16 + g*8 + i]   i = 0..7
// B (32x16, KxN) stored in LDS transposed as Bs[n][k]: lane L -> n = L%16.
//   b[i]   = B[kbase + g*8 + i][n], b[8+i] = B[kbase + 16 + g*8 + i][n]  (i = 0..7)
union FragU { uint4 u[2]; v16h v; };

__device__ __forceinline__ v16h lds_load_a(const _Float16* rowbase, int kbase, int g) {
    FragU r;
    r.u[0] = *reinterpret_cast<const uint4*>(rowbase + kbase + g * 8);
    r.u[1] = *reinterpret_cast<const uint4*>(rowbase + kbase + 16 + g * 8);
    return r.v;
}

__device__ __forceinline__ v16h lds_load_b(const _Float16* rowbase, int kbase, int g) {
    FragU r;
    r.u[0] = *reinterpret_cast<const uint4*>(rowbase + kbase + g * 8);
    r.u[1] = *reinterpret_cast<const uint4*>(rowbase + kbase + 16 + g * 8);
    return r.v;
}

__device__ __forceinline__ v8f zero_v8f() {
    v8f z;
#pragma unroll
    for (int i = 0; i < 8; ++i) z[i] = 0.0f;
    return z;
}

// ---------------- weight convert (f32 -> f16, layout preserved: (Nout, K)) ----------------
__global__ __launch_bounds__(256) void f32_to_f16_kernel(const float* __restrict__ src,
                                                         _Float16* __restrict__ dst, int n) {
    int i = blockIdx.x * 256 + threadIdx.x;
    if (i < n) dst[i] = (_Float16)src[i];
}

// ---------------- adaLN conditioning: ada = silu(c) @ ada_w^T + ada_b ----------------
__global__ __launch_bounds__(256) void ada_kernel(const float* __restrict__ c,
                                                  const float* __restrict__ w,
                                                  const float* __restrict__ b,
                                                  float* __restrict__ ada) {
    __shared__ float sc[CDIM];
    const int n = blockIdx.y;
    const int j = blockIdx.x * 256 + threadIdx.x;     // 0..4607
    for (int k = threadIdx.x; k < CDIM; k += 256) {
        float v = c[n * CDIM + k];
        sc[k] = v / (1.0f + __expf(-v));
    }
    __syncthreads();
    float acc = b[j];
    const float* wr = w + (size_t)j * CDIM;
#pragma unroll 4
    for (int k = 0; k < CDIM; ++k) acc += sc[k] * wr[k];
    ada[n * 6 * CDIM + j] = acc;
}

// ---------------- LayerNorm + modulate -> f16 (A operand for GEMM) ----------------
__global__ __launch_bounds__(256) void ln_mod_kernel(const float* __restrict__ x,
                                                     const float* __restrict__ ada,
                                                     _Float16* __restrict__ h,
                                                     int shOff, int scOff, int winMap) {
    __shared__ float r1[256], r2[256];
    const int row = blockIdx.x;           // 0..32767
    const int tid = threadIdx.x;

    int n;
    const float* xr;
    if (winMap) {
        int b_ = row >> 6, t = row & 63;
        n = b_ >> 6;
        int win = b_ & 63;
        int gh = ((win >> 3) << 3) + (t >> 3);
        int gw = ((win & 7) << 3) + (t & 7);
        int oh = (gh + SHIFTC) & 63;
        int ow = (gw + SHIFTC) & 63;
        xr = x + ((size_t)n * TSEQ + (oh << 6) + ow) * CDIM;
    } else {
        n = row >> 12;
        xr = x + (size_t)row * CDIM;
    }

    float x0 = xr[tid], x1 = xr[tid + 256], x2 = xr[tid + 512];
    r1[tid] = x0 + x1 + x2;
    r2[tid] = x0 * x0 + x1 * x1 + x2 * x2;
    __syncthreads();
#pragma unroll
    for (int s = 128; s > 0; s >>= 1) {
        if (tid < s) { r1[tid] += r1[tid + s]; r2[tid] += r2[tid + s]; }
        __syncthreads();
    }
    const float mean = r1[0] * (1.0f / (float)CDIM);
    const float var  = r2[0] * (1.0f / (float)CDIM) - mean * mean;
    const float rs   = rsqrtf(var + 1e-6f);

    const float* shp = ada + n * 6 * CDIM + shOff;
    const float* scp = ada + n * 6 * CDIM + scOff;
    _Float16* hr = h + (size_t)row * CDIM;
    hr[tid]       = (_Float16)((x0 - mean) * rs * (1.0f + scp[tid])       + shp[tid]);
    hr[tid + 256] = (_Float16)((x1 - mean) * rs * (1.0f + scp[tid + 256]) + shp[tid + 256]);
    hr[tid + 512] = (_Float16)((x2 - mean) * rs * (1.0f + scp[tid + 512]) + shp[tid + 512]);
}

// ---------------- generic WMMA GEMM: C = A(MxK,f16) * B(NxK,f16)^T + bias ----------------
// Double-buffered LDS (dynamic, 72KB) + async global->LDS staging.
// epi: 0=f16 out, 1=f16 out with tanh-GELU, 2=proj residual (window unpartition +
//      inverse shift + gated residual, f32 out), 3=fc2 final gated residual (f32 out)
constexpr int LDK   = 72;           // 64 + 8 halves pad: conflict-free, 16B aligned rows
constexpr int TILEH = 128 * LDK;    // halves per tile buffer
#define GEMM_LDS_BYTES (4 * TILEH * 2)   // 2 bufs x (A + B) x f16

__device__ __forceinline__ void stage_tile(const _Float16* __restrict__ A,
                                           const _Float16* __restrict__ B,
                                           _Float16* As, _Float16* Bs,
                                           int m0, int n0, int K, int kt, int tid) {
#pragma unroll
    for (int c2 = 0; c2 < 4; ++c2) {
        int idx = tid + c2 * 256;          // 0..1023 chunks of 8 halves
        int row = idx >> 3, part = idx & 7;
        cp16_async(A + (size_t)(m0 + row) * K + kt + part * 8, &As[row * LDK + part * 8]);
        cp16_async(B + (size_t)(n0 + row) * K + kt + part * 8, &Bs[row * LDK + part * 8]);
    }
}

__global__ __launch_bounds__(256)
void wmma_gemm_kernel(const _Float16* __restrict__ A,
                      const _Float16* __restrict__ B,
                      const float* __restrict__ bias,
                      _Float16* __restrict__ outh,
                      float* __restrict__ outf,
                      const float* __restrict__ resid,
                      const float* __restrict__ ada,
                      int adaOff, int M, int Nn, int K, int epi) {
    extern __shared__ char smem[];
    _Float16* AsBase = reinterpret_cast<_Float16*>(smem);
    _Float16* BsBase = AsBase + 2 * TILEH;

    const int tid  = threadIdx.x;
    const int lane = tid & 31;
    const int g    = lane >> 4;
    const int wv   = tid >> 5;        // 8 waves
    const int wm   = wv & 3;          // 4 along M (32 rows each)
    const int wn   = wv >> 2;         // 2 along N (64 cols each)
    const int m0   = blockIdx.y * 128;
    const int n0   = blockIdx.x * 128;

    v8f acc[2][4];
#pragma unroll
    for (int i = 0; i < 2; ++i)
#pragma unroll
        for (int j = 0; j < 4; ++j) acc[i][j] = zero_v8f();

    const int KT = K >> 6;            // K tiles of 64

    stage_tile(A, B, AsBase, BsBase, m0, n0, K, 0, tid);
    cp_wait_all();
    __syncthreads();

    for (int kt = 0; kt < KT; ++kt) {
        const int cur = kt & 1;
        const _Float16* Asc = AsBase + cur * TILEH;
        const _Float16* Bsc = BsBase + cur * TILEH;

        // prefetch next tile into the other buffer while computing this one
        if (kt + 1 < KT)
            stage_tile(A, B, AsBase + (1 - cur) * TILEH, BsBase + (1 - cur) * TILEH,
                       m0, n0, K, (kt + 1) * 64, tid);

#pragma unroll
        for (int ks = 0; ks < 64; ks += 32) {
            v16h af[2], bf[4];
#pragma unroll
            for (int i = 0; i < 2; ++i)
                af[i] = lds_load_a(&Asc[(wm * 32 + i * 16 + (lane & 15)) * LDK], ks, g);
#pragma unroll
            for (int j = 0; j < 4; ++j)
                bf[j] = lds_load_b(&Bsc[(wn * 64 + j * 16 + (lane & 15)) * LDK], ks, g);
#pragma unroll
            for (int i = 0; i < 2; ++i)
#pragma unroll
                for (int j = 0; j < 4; ++j)
                    acc[i][j] = __builtin_amdgcn_wmma_f32_16x16x32_f16(
                        false, af[i], false, bf[j], (short)0, acc[i][j], false, false);
        }

        if (kt + 1 < KT) cp_wait_all();
        __syncthreads();
    }

    // epilogue; C layout: element r of acc at lane L = C[m = r + 8*(L/16)][n = L%16]
#pragma unroll
    for (int i = 0; i < 2; ++i) {
#pragma unroll
        for (int j = 0; j < 4; ++j) {
            const int col     = n0 + wn * 64 + j * 16 + (lane & 15);
            const int rowBase = m0 + wm * 32 + i * 16 + (g << 3);
            const float bv    = bias[col];
#pragma unroll
            for (int r = 0; r < 8; ++r) {
                const int row = rowBase + r;
                float v = acc[i][j][r] + bv;
                if (epi == 0) {
                    outh[(size_t)row * Nn + col] = (_Float16)v;
                } else if (epi == 1) {
                    float t = 0.7978845608f * (v + 0.044715f * v * v * v);
                    float gl = 0.5f * v * (1.0f + tanhf(t));
                    outh[(size_t)row * Nn + col] = (_Float16)gl;
                } else if (epi == 2) {
                    // row = window-token -> (n, oh, ow) after inverse cyclic shift
                    int b_ = row >> 6, t = row & 63;
                    int n  = b_ >> 6, win = b_ & 63;
                    int gh = ((win >> 3) << 3) + (t >> 3);
                    int gw = ((win & 7) << 3) + (t & 7);
                    int oh = (gh + SHIFTC) & 63;
                    int ow = (gw + SHIFTC) & 63;
                    size_t idx = ((size_t)n * TSEQ + (oh << 6) + ow) * CDIM + col;
                    float gate = ada[n * 6 * CDIM + adaOff + col];
                    outf[idx] = resid[idx] + gate * v;
                } else {
                    int n = row >> 12;
                    size_t idx = (size_t)row * CDIM + col;
                    float gate = ada[n * 6 * CDIM + adaOff + col];
                    outf[idx] = resid[idx] + gate * v;
                }
            }
        }
    }
}

// ---------------- window attention: one block per (window, head) ----------------
__global__ __launch_bounds__(128)
void attn_kernel(const _Float16* __restrict__ qkv, _Float16* __restrict__ ob) {
    constexpr int LD = 72;
    __shared__ alignas(16) _Float16 Qs[TW * LD];
    __shared__ alignas(16) _Float16 Ks[TW * LD];
    __shared__ alignas(16) _Float16 Vt[HD * LD];   // V transposed: [d][t]
    __shared__ alignas(16) _Float16 Ps[TW * LD];
    __shared__ float Ss[TW * 67];                  // stride 67 -> conflict-free rows

    const int tid  = threadIdx.x;
    const int lane = tid & 31;
    const int g    = lane >> 4;
    const int wv   = tid >> 5;                     // 4 waves, 16 rows each
    const int blk  = blockIdx.x;
    const int b_   = blk / NHEAD;
    const int hh   = blk % NHEAD;
    const size_t base = (size_t)b_ * TW * (3 * CDIM) + hh * HD;

    // stage Q, K (row-major [t][d]) via async global->LDS, 16B chunks
#pragma unroll
    for (int c2 = 0; c2 < 4; ++c2) {
        int idx = tid + c2 * 128;                  // 0..511
        int t = idx >> 3, part = idx & 7;
        cp16_async(qkv + base + (size_t)t * (3 * CDIM) + part * 8, &Qs[t * LD + part * 8]);
        cp16_async(qkv + base + (size_t)t * (3 * CDIM) + CDIM + part * 8, &Ks[t * LD + part * 8]);
    }
    // stage V transposed so P@V B-operand reads are contiguous per lane
    for (int e = tid; e < TW * HD; e += 128) {
        int t = e >> 6, d = e & 63;
        Vt[d * LD + t] = qkv[base + (size_t)t * (3 * CDIM) + 2 * CDIM + d];
    }
    cp_wait_all();
    __syncthreads();

    // S = Q @ K^T * (1/sqrt(hd)); wave wv owns rows [wv*16, wv*16+16)
#pragma unroll
    for (int nt = 0; nt < 4; ++nt) {
        v8f acc = zero_v8f();
#pragma unroll
        for (int ks = 0; ks < HD; ks += 32) {
            v16h a = lds_load_a(&Qs[(wv * 16 + (lane & 15)) * LD], ks, g);
            v16h b = lds_load_b(&Ks[(nt * 16 + (lane & 15)) * LD], ks, g);
            acc = __builtin_amdgcn_wmma_f32_16x16x32_f16(false, a, false, b, (short)0, acc, false, false);
        }
        const int col = nt * 16 + (lane & 15);
        const int rb  = wv * 16 + (g << 3);
#pragma unroll
        for (int r = 0; r < 8; ++r) Ss[(rb + r) * 67 + col] = acc[r] * 0.125f;
    }
    __syncthreads();

    // softmax: one thread per row
    if (tid < TW) {
        float* srow = &Ss[tid * 67];
        float mx = srow[0];
#pragma unroll 8
        for (int j = 1; j < TW; ++j) mx = fmaxf(mx, srow[j]);
        float sum = 0.0f;
#pragma unroll 8
        for (int j = 0; j < TW; ++j) { float e = __expf(srow[j] - mx); srow[j] = e; sum += e; }
        float inv = 1.0f / sum;
#pragma unroll 8
        for (int j = 0; j < TW; ++j) Ps[tid * LD + j] = (_Float16)(srow[j] * inv);
    }
    __syncthreads();

    // O = P @ V -> write f16 to (B_, Tw, C) head slice
#pragma unroll
    for (int nt = 0; nt < 4; ++nt) {
        v8f acc = zero_v8f();
#pragma unroll
        for (int ks = 0; ks < TW; ks += 32) {
            v16h a = lds_load_a(&Ps[(wv * 16 + (lane & 15)) * LD], ks, g);
            v16h b = lds_load_b(&Vt[(nt * 16 + (lane & 15)) * LD], ks, g);
            acc = __builtin_amdgcn_wmma_f32_16x16x32_f16(false, a, false, b, (short)0, acc, false, false);
        }
        const int col = hh * HD + nt * 16 + (lane & 15);
        const int rb  = wv * 16 + (g << 3);
#pragma unroll
        for (int r = 0; r < 8; ++r)
            ob[(size_t)(b_ * TW + rb + r) * CDIM + col] = (_Float16)acc[r];
    }
}

// ---------------- launcher ----------------
extern "C" void kernel_launch(void* const* d_in, const int* in_sizes, int n_in,
                              void* d_out, int out_size, void* d_ws, size_t ws_size,
                              hipStream_t stream) {
    (void)in_sizes; (void)n_in; (void)out_size; (void)ws_size;
    const float* x_seq  = (const float*)d_in[0];
    const float* cvec   = (const float*)d_in[1];
    const float* qkv_w  = (const float*)d_in[2];
    const float* qkv_b  = (const float*)d_in[3];
    const float* proj_w = (const float*)d_in[4];
    const float* proj_b = (const float*)d_in[5];
    const float* fc1_w  = (const float*)d_in[6];
    const float* fc1_b  = (const float*)d_in[7];
    const float* fc2_w  = (const float*)d_in[8];
    const float* fc2_b  = (const float*)d_in[9];
    const float* ada_w  = (const float*)d_in[10];
    const float* ada_b  = (const float*)d_in[11];

    char* ws = (char*)d_ws;
    auto alloc = [&](size_t bytes) -> char* {
        char* p = ws;
        ws += (bytes + 255) & ~(size_t)255;
        return p;
    };
    float*    ada  = (float*)   alloc((size_t)NIMG * 6 * CDIM * 4);
    _Float16* wq   = (_Float16*)alloc((size_t)3 * CDIM * CDIM * 2);
    _Float16* wp   = (_Float16*)alloc((size_t)CDIM * CDIM * 2);
    _Float16* w1   = (_Float16*)alloc((size_t)HIDD * CDIM * 2);
    _Float16* w2   = (_Float16*)alloc((size_t)CDIM * HIDD * 2);
    _Float16* hbuf = (_Float16*)alloc((size_t)MROWS * CDIM * 2);
    _Float16* qkvb = (_Float16*)alloc((size_t)MROWS * 3 * CDIM * 2);
    _Float16* attb = (_Float16*)alloc((size_t)MROWS * CDIM * 2);
    float*    xmid = (float*)   alloc((size_t)MROWS * CDIM * 4);
    _Float16* f1o  = (_Float16*)alloc((size_t)MROWS * HIDD * 2);

    // weight conversions (f32 -> f16, (Nout,K) layout preserved)
    f32_to_f16_kernel<<<(3 * CDIM * CDIM + 255) / 256, 256, 0, stream>>>(qkv_w, wq, 3 * CDIM * CDIM);
    f32_to_f16_kernel<<<(CDIM * CDIM + 255) / 256, 256, 0, stream>>>(proj_w, wp, CDIM * CDIM);
    f32_to_f16_kernel<<<(HIDD * CDIM + 255) / 256, 256, 0, stream>>>(fc1_w, w1, HIDD * CDIM);
    f32_to_f16_kernel<<<(CDIM * HIDD + 255) / 256, 256, 0, stream>>>(fc2_w, w2, CDIM * HIDD);

    // adaLN conditioning
    ada_kernel<<<dim3(6 * CDIM / 256, NIMG), 256, 0, stream>>>(cvec, ada_w, ada_b, ada);

    // LN + modulate (MSA), folding shift + window partition into addressing
    ln_mod_kernel<<<MROWS, 256, 0, stream>>>(x_seq, ada, hbuf, 0 * CDIM, 1 * CDIM, 1);

    // QKV GEMM: (32768 x 768) @ (768 x 2304)
    wmma_gemm_kernel<<<dim3(3 * CDIM / 128, MROWS / 128), 256, GEMM_LDS_BYTES, stream>>>(
        hbuf, wq, qkv_b, qkvb, nullptr, nullptr, ada, 0, MROWS, 3 * CDIM, CDIM, 0);

    // window attention
    attn_kernel<<<BWIN * NHEAD, 128, 0, stream>>>(qkvb, attb);

    // proj GEMM + window unpartition + inverse shift + gated residual -> xmid (f32)
    wmma_gemm_kernel<<<dim3(CDIM / 128, MROWS / 128), 256, GEMM_LDS_BYTES, stream>>>(
        attb, wp, proj_b, nullptr, xmid, x_seq, ada, 2 * CDIM, MROWS, CDIM, CDIM, 2);

    // LN + modulate (MLP)
    ln_mod_kernel<<<MROWS, 256, 0, stream>>>(xmid, ada, hbuf, 3 * CDIM, 4 * CDIM, 0);

    // fc1 GEMM + GELU
    wmma_gemm_kernel<<<dim3(HIDD / 128, MROWS / 128), 256, GEMM_LDS_BYTES, stream>>>(
        hbuf, w1, fc1_b, f1o, nullptr, nullptr, ada, 0, MROWS, HIDD, CDIM, 1);

    // fc2 GEMM + gated residual -> final output (f32)
    wmma_gemm_kernel<<<dim3(CDIM / 128, MROWS / 128), 256, GEMM_LDS_BYTES, stream>>>(
        f1o, w2, fc2_b, nullptr, (float*)d_out, xmid, ada, 5 * CDIM, MROWS, CDIM, HIDD, 3);
}
